// RecurrentNetwork_53077205844331
// MI455X (gfx1250) — matrix operand order repaired and checked by reading, not compiled, not run on hardware
//
#include <hip/hip_runtime.h>

// ---------------------------------------------------------------------------
// Types / helpers
// ---------------------------------------------------------------------------
typedef unsigned short u16;
typedef __attribute__((ext_vector_type(16))) __bf16 v16bf;
typedef __attribute__((ext_vector_type(8)))  float  v8f;

union Frag { uint4 q[2]; v16bf v; };

__device__ __forceinline__ u16 f2bf(float f) {
  unsigned u = __float_as_uint(f);
  u += 0x7FFFu + ((u >> 16) & 1u);          // round-to-nearest-even
  return (u16)(u >> 16);
}
__device__ __forceinline__ float bf2f(u16 h) {
  return __uint_as_float(((unsigned)h) << 16);
}

static constexpr int kS  = 512;   // sequence length
static constexpr int kB  = 128;   // batch
static constexpr int kE  = 300;   // embedding dim
static constexpr int kH  = 150;   // hidden dim
static constexpr int kHP = 160;   // hidden dim padded to 10 * 16

// ---------------------------------------------------------------------------
// Weight pad+convert: [rows, cols] f32 -> [160, kpad] bf16 (zero padded)
// ---------------------------------------------------------------------------
__global__ void pad_weight_kernel(const float* __restrict__ w, u16* __restrict__ o,
                                  int rows, int cols, int kpad) {
  int i = blockIdx.x * 256 + threadIdx.x;
  int total = kHP * kpad;
  if (i >= total) return;
  int n = i / kpad, k = i % kpad;
  float v = (n < rows && k < cols) ? w[n * cols + k] : 0.0f;
  o[i] = f2bf(v);
}

__global__ void combine_bias_kernel(const float* __restrict__ a,
                                    const float* __restrict__ b,
                                    float* __restrict__ o) {
  int i = threadIdx.x;
  if (i < kHP) o[i] = (i < kH) ? (a[i] + b[i]) : 0.0f;
}

__global__ void zero_u16_kernel(u16* __restrict__ p, int n) {
  int i = blockIdx.x * 256 + threadIdx.x;
  if (i < n) p[i] = 0;
}

// ---------------------------------------------------------------------------
// One tanh-RNN layer, persistent over all 512 timesteps.
//   KI    = number of K-tiles of 32 for the input GEMM (10 for Din=320, 5 for 160)
//   EMBED = fuse embedding-table gather into the x staging (layer 0 only)
//
// Block: 320 threads = 10 waves; wave w owns output columns [16w, 16w+16).
// Grid : 8 blocks, each a 16-row batch tile (recurrence is per-row independent).
// Weights are WMMA B-fragments held in VGPRs across the whole t-loop.
// x_{t+1} is software-pipelined: global loads issued before the WMMA chain of
// step t, committed to the other LDS buffer after the epilogue -> global load
// latency hidden behind compute; single barrier per timestep.
// ---------------------------------------------------------------------------
template <int KI, bool EMBED>
__global__ __launch_bounds__(320)
void rnn_layer_kernel(const int*   __restrict__ tokens,   // [B,S]      (EMBED)
                      const float* __restrict__ emb,      // [V,E]      (EMBED)
                      const u16*   __restrict__ xin,      // [S,B,KPAD] (!EMBED)
                      const u16*   __restrict__ wih,      // [160,KPAD] bf16
                      const u16*   __restrict__ whh,      // [160,160]  bf16
                      const float* __restrict__ bias,     // [160] combined
                      const u16*   __restrict__ h0,       // [B,160] bf16
                      u16*         __restrict__ y,        // [S,B,160] bf16
                      u16*         __restrict__ hfin) {   // [B,160] bf16
  constexpr int KPAD = KI * 32;
  constexpr int XPT  = (16 * KPAD / 8) / 320;   // 8-elem chunks per thread (2 or 1)
  __shared__ __align__(16) u16 x_lds[2][16 * KPAD];
  __shared__ __align__(16) u16 h_lds[2][16 * kHP];

  const int tid   = threadIdx.x;
  const int wave  = tid >> 5;        // 0..9 -> N tile
  const int lane  = tid & 31;
  const int half  = lane >> 4;
  const int lrow  = lane & 15;
  const int bbase = blockIdx.x * 16; // batch tile base row

  // ---- preload loop-invariant B (weight) fragments into registers --------
  // B[k][n] = W[n][k]; lane layout: n = lane&15, half 0 -> K {0..7,16..23}; half 1 -> K {8..15,24..31}
  v16bf wihf[KI];
  v16bf whhf[5];
  {
    const int n  = wave * 16 + lrow;
    const int kr = half * 8;
#pragma unroll
    for (int kt = 0; kt < KI; ++kt) {
      Frag f;
      const uint4* p = (const uint4*)(wih + n * KPAD + kt * 32 + kr);
      f.q[0] = p[0]; f.q[1] = p[2];
      wihf[kt] = f.v;
    }
#pragma unroll
    for (int kt = 0; kt < 5; ++kt) {
      Frag f;
      const uint4* p = (const uint4*)(whh + n * kHP + kt * 32 + kr);
      f.q[0] = p[0]; f.q[1] = p[2];
      whhf[kt] = f.v;
    }
  }
  const float bn = bias[wave * 16 + lrow];  // C layout: lane owns fixed column

  // ---- x pipeline registers ----------------------------------------------
  float fr[XPT][8];   // EMBED: raw f32 gathered from the embedding table
  uint4 ur[XPT];      // !EMBED: raw bf16x8 chunks

  // issue global loads for timestep tt (no waiting here)
  auto fetch_x = [&](int tt) {
    if (EMBED) {
#pragma unroll
      for (int c = 0; c < XPT; ++c) {
        int idx = (tid + c * 320) * 8;
        int r = idx / KPAD, col0 = idx % KPAD;
        int tok = tokens[(bbase + r) * kS + tt];
        const float* er = emb + (size_t)tok * kE;
#pragma unroll
        for (int e = 0; e < 8; ++e) {
          int col = col0 + e;
          fr[c][e] = (col < kE) ? er[col] : 0.0f;
        }
      }
    } else {
      const u16* src = xin + ((size_t)tt * kB + bbase) * KPAD;
#pragma unroll
      for (int c = 0; c < XPT; ++c) {
        int idx = (tid + c * 320) * 8;
        ur[c] = *(const uint4*)(src + idx);
        if (tt + 1 < kS)                      // reach ahead one more step
          __builtin_prefetch(src + (size_t)kB * KPAD + idx, 0, 1);
      }
    }
  };
  // commit pipeline registers into LDS x buffer `buf` (bf16 convert for EMBED)
  auto commit_x = [&](int buf) {
#pragma unroll
    for (int c = 0; c < XPT; ++c) {
      int idx = (tid + c * 320) * 8;
      if (EMBED) {
        u16 tmp[8];
#pragma unroll
        for (int e = 0; e < 8; ++e) tmp[e] = f2bf(fr[c][e]);
        *(uint4*)&x_lds[buf][idx] = *(uint4*)tmp;
      } else {
        *(uint4*)&x_lds[buf][idx] = ur[c];
      }
    }
  };

  // ---- initial hidden state into LDS buffer 0 + stage x_0 ----------------
  for (int i = tid; i < (16 * kHP) / 8; i += 320) {
    int idx = i * 8;
    int r = idx / kHP, c = idx % kHP;
    *(uint4*)&h_lds[0][idx] = *(const uint4*)(h0 + (bbase + r) * kHP + c);
  }
  fetch_x(0);
  commit_x(0);
  __syncthreads();

  int xcur = 0, hcur = 0;
  for (int t = 0; t < kS; ++t) {
    // ---- issue next timestep's global loads (latency hidden by WMMAs) ---
    if (t + 1 < kS) fetch_x(t + 1);

    // ---- C = x_t * Wih^T + h * Whh^T  (f32 accum over bf16 WMMA) --------
    v8f acc = {};
#pragma unroll
    for (int kt = 0; kt < KI; ++kt) {
      // A layout: half 0 -> K {0..7,16..23}; half 1 -> K {8..15,24..31}
      Frag a;
      const uint4* p = (const uint4*)&x_lds[xcur][lrow * KPAD + kt * 32 + half * 8];
      a.q[0] = p[0]; a.q[1] = p[2];   // +32 bytes = +16 elements
      acc = __builtin_amdgcn_wmma_f32_16x16x32_bf16(
          false, a.v, false, wihf[kt], (short)0, acc, false, false);
    }
#pragma unroll
    for (int kt = 0; kt < 5; ++kt) {
      Frag a;
      const uint4* p = (const uint4*)&h_lds[hcur][lrow * kHP + kt * 32 + half * 8];
      a.q[0] = p[0]; a.q[1] = p[2];
      acc = __builtin_amdgcn_wmma_f32_16x16x32_bf16(
          false, a.v, false, whhf[kt], (short)0, acc, false, false);
    }

    // ---- epilogue: bias + tanh, write new h tile ------------------------
    const int hn   = hcur ^ 1;
    const int ncol = wave * 16 + lrow;   // lane's output column
#pragma unroll
    for (int i = 0; i < 8; ++i) {
      int m = i + half * 8;              // lane's output row for VGPR i
      float v = tanhf(acc[i] + bn);
      h_lds[hn][m * kHP + ncol] = f2bf(v);
    }

    // ---- commit x_{t+1} into the other x buffer (waits its loads) -------
    if (t + 1 < kS) commit_x(xcur ^ 1);

    __syncthreads();  // h_lds[hn] + x_lds[xcur^1] complete; old buffers free

    // ---- y_t (and final hidden) out to global, coalesced b128 -----------
    {
      u16* dst = y + ((size_t)t * kB + bbase) * kHP;
      for (int i = tid; i < (16 * kHP) / 8; i += 320) {
        int idx = i * 8;
        int r = idx / kHP, c = idx % kHP;
        uint4 q = *(uint4*)&h_lds[hn][idx];
        *(uint4*)(dst + r * kHP + c) = q;
        if (t == kS - 1) *(uint4*)(hfin + (bbase + r) * kHP + c) = q;
      }
    }
    xcur ^= 1;
    hcur  = hn;
  }
}

// ---------------------------------------------------------------------------
// Head: out[b] = relu(last[b] @ w1^T + b1) @ w2^T + b2   (5.8 MFLOP, VALU ok)
// ---------------------------------------------------------------------------
__global__ void head_kernel(const u16* __restrict__ ylast,   // [B,160] bf16
                            const float* __restrict__ w1,    // [150,150]
                            const float* __restrict__ b1,    // [150]
                            const float* __restrict__ w2,    // [1,150]
                            const float* __restrict__ b2,    // [1]
                            float* __restrict__ out) {       // [B,1]
  int b = blockIdx.x * blockDim.x + threadIdx.x;
  if (b >= kB) return;
  const u16* xr = ylast + b * kHP;
  float acc = 0.0f;
  for (int j = 0; j < kH; ++j) {
    float s = b1[j];
    const float* wr = w1 + j * kH;
    for (int k = 0; k < kH; ++k) s = fmaf(bf2f(xr[k]), wr[k], s);
    acc = fmaf(fmaxf(s, 0.0f), w2[j], acc);
  }
  out[b] = acc + b2[0];
}

// ---------------------------------------------------------------------------
// Launch
// ---------------------------------------------------------------------------
extern "C" void kernel_launch(void* const* d_in, const int* in_sizes, int n_in,
                              void* d_out, int out_size, void* d_ws, size_t ws_size,
                              hipStream_t stream) {
  (void)in_sizes; (void)n_in; (void)out_size; (void)ws_size;

  const int*   tokens = (const int*)d_in[0];
  const float* emb    = (const float*)d_in[1];
  // per-tag weights: w_ih, w_hh, b_ih, b_hh for tags 00,01,10,11
  const float* w_ih[4], *w_hh[4], *b_ih[4], *b_hh[4];
  for (int l = 0; l < 4; ++l) {
    w_ih[l] = (const float*)d_in[2 + 4 * l + 0];
    w_hh[l] = (const float*)d_in[2 + 4 * l + 1];
    b_ih[l] = (const float*)d_in[2 + 4 * l + 2];
    b_hh[l] = (const float*)d_in[2 + 4 * l + 3];
  }
  const float* w1 = (const float*)d_in[18];
  const float* b1 = (const float*)d_in[19];
  const float* w2 = (const float*)d_in[20];
  const float* b2 = (const float*)d_in[21];
  float* out = (float*)d_out;

  // ---- workspace carve-out (aligned 256B) --------------------------------
  char* ws = (char*)d_ws;
  size_t off = 0;
  auto take = [&](size_t bytes) -> void* {
    void* p = ws + off;
    off = (off + bytes + 255) & ~(size_t)255;
    return p;
  };
  u16* wihp[4]; u16* whhp[4]; float* biasc[4];
  wihp[0] = (u16*)take((size_t)kHP * 320 * 2);                  // layer0: Din pad 320
  for (int l = 1; l < 4; ++l) wihp[l] = (u16*)take((size_t)kHP * kHP * 2);
  for (int l = 0; l < 4; ++l) whhp[l] = (u16*)take((size_t)kHP * kHP * 2);
  for (int l = 0; l < 4; ++l) biasc[l] = (float*)take((size_t)kHP * 4);
  u16* h0z  = (u16*)take((size_t)kB * kHP * 2);
  u16* hf00 = (u16*)take((size_t)kB * kHP * 2);
  u16* hf01 = (u16*)take((size_t)kB * kHP * 2);
  u16* hft  = (u16*)take((size_t)kB * kHP * 2);
  u16* y0   = (u16*)take((size_t)kS * kB * kHP * 2);
  u16* y1   = (u16*)take((size_t)kS * kB * kHP * 2);

  // ---- weight / bias prep -----------------------------------------------
  pad_weight_kernel<<<(kHP * 320 + 255) / 256, 256, 0, stream>>>(w_ih[0], wihp[0], kH, kE, 320);
  for (int l = 1; l < 4; ++l)
    pad_weight_kernel<<<(kHP * kHP + 255) / 256, 256, 0, stream>>>(w_ih[l], wihp[l], kH, kH, kHP);
  for (int l = 0; l < 4; ++l)
    pad_weight_kernel<<<(kHP * kHP + 255) / 256, 256, 0, stream>>>(w_hh[l], whhp[l], kH, kH, kHP);
  for (int l = 0; l < 4; ++l)
    combine_bias_kernel<<<1, kHP, 0, stream>>>(b_ih[l], b_hh[l], biasc[l]);
  zero_u16_kernel<<<(kB * kHP + 255) / 256, 256, 0, stream>>>(h0z, kB * kHP);

  // ---- four RNN layers (batch tiled 16 rows/block; 10 waves/block) -------
  dim3 grid(kB / 16), block(320);
  // layer 00: fused embedding gather, h0 = 0
  rnn_layer_kernel<10, true><<<grid, block, 0, stream>>>(
      tokens, emb, nullptr, wihp[0], whhp[0], biasc[0], h0z, y0, hf00);
  // layer 01: input y0, h0 = 0
  rnn_layer_kernel<5, false><<<grid, block, 0, stream>>>(
      nullptr, nullptr, y0, wihp[1], whhp[1], biasc[1], h0z, y1, hf01);
  // layer 10: input y1, h0 = h_final(layer 00)
  rnn_layer_kernel<5, false><<<grid, block, 0, stream>>>(
      nullptr, nullptr, y1, wihp[2], whhp[2], biasc[2], hf00, y0, hft);
  // layer 11: input y0, h0 = h_final(layer 01)
  rnn_layer_kernel<5, false><<<grid, block, 0, stream>>>(
      nullptr, nullptr, y0, wihp[3], whhp[3], biasc[3], hf01, y1, hft);

  // ---- head on last timestep of layer 11 ---------------------------------
  head_kernel<<<1, kB, 0, stream>>>(y1 + (size_t)(kS - 1) * kB * kHP, w1, b1, w2, b2, out);
}
